// MoDInfiniTransformer_21895743275675
// MI455X (gfx1250) — compile-verified, matrix operand order checked by reading, not run on hardware
//
#include <hip/hip_runtime.h>

// ---------------------------------------------------------------------------
// MoD + Infini-attention block for MI455X (gfx1250, wave32, WMMA).
// All matmuls lowered to v_wmma_f32_16x16x32_bf16; LDS layouts chosen so every
// hot fragment load is 2x ds_load_b128 (contiguous per-lane runs).
// ---------------------------------------------------------------------------

typedef __attribute__((ext_vector_type(16))) __bf16 v16bf;
typedef __attribute__((ext_vector_type(8)))  float  v8f;

// problem constants
constexpr int Bc   = 4;
constexpr int Sc   = 8192;
constexpr int Dc   = 2048;
constexpr int Hc   = 16;
constexpr int DKc  = 128;
constexpr int DVc  = 128;
constexpr int DHc  = 8192;
constexpr int SEGc = 256;             // routed tokens per full segment
constexpr int FSEG = 2048;            // full segment
constexpr int NSF  = Sc / FSEG;       // 4 full segments
constexpr int RLc  = NSF * SEGc;      // 1024 routed tokens per batch
constexpr int MSEL = Bc * RLc;        // 4096 routed rows total

// ---------------------------------------------------------------------------
// helpers
// ---------------------------------------------------------------------------
__device__ inline __bf16 f2bf(float f) {
    unsigned u; __builtin_memcpy(&u, &f, 4);
    unsigned r = (u + 0x7FFFu + ((u >> 16) & 1u)) >> 16;
    unsigned short h = (unsigned short)r;
    __bf16 b; __builtin_memcpy(&b, &h, 2);
    return b;
}
__device__ inline float bf2f(__bf16 b) {
    unsigned short h; __builtin_memcpy(&h, &b, 2);
    unsigned u = ((unsigned)h) << 16;
    float f; __builtin_memcpy(&f, &u, 4);
    return f;
}
__device__ inline float elu1(float x) { return x > 0.0f ? x + 1.0f : __expf(x); }
__device__ inline v8f vzero() {
    v8f z;
#pragma unroll
    for (int r = 0; r < 8; ++r) z[r] = 0.0f;
    return z;
}
__device__ inline v8f wmma_bf16(v16bf a, v16bf b, v8f c) {
    return __builtin_amdgcn_wmma_f32_16x16x32_bf16(
        false, a, false, b, (short)0, c, false, false);
}

union frag_u { uint4 u[2]; v16bf v; };

// Fragment from a [i][k] tile with k innermost, 32-wide (A tile or transposed
// B tile).  Lane layout (ISA 7.12.2): lane<16 -> K = {kb..kb+7, kb+16..kb+23}
// with kb = 8*(lane>=16).  Both 8-element runs are contiguous -> 2x b128.
__device__ inline v16bf frag32(const __bf16* base, int i0) {
    int lane = threadIdx.x & 31;
    int i  = i0 + (lane & 15);
    int kb = (lane >> 4) << 3;
    frag_u c;
    c.u[0] = *(const uint4*)(base + i * 32 + kb);
    c.u[1] = *(const uint4*)(base + i * 32 + 16 + kb);
    return c.v;
}
// B fragment where B[k][n] = S[(row0+n)*128 + d0 + k]  (k^T: scores = q k^T)
__device__ inline v16bf fragKT(const __bf16* S, int row0, int d0) {
    int lane = threadIdx.x & 31;
    int n  = lane & 15;
    int kb = (lane >> 4) << 3;
    const __bf16* p = S + (row0 + n) * 128 + d0;
    frag_u c;
    c.u[0] = *(const uint4*)(p + kb);
    c.u[1] = *(const uint4*)(p + 16 + kb);
    return c.v;
}
// B fragment where B[k][n] = T[(d0+n)*256 + row0 + k]  (transposed V tile)
__device__ inline v16bf frag256(const __bf16* T, int d0, int row0) {
    int lane = threadIdx.x & 31;
    int n  = lane & 15;
    int kb = (lane >> 4) << 3;
    const __bf16* p = T + (d0 + n) * 256 + row0;
    frag_u c;
    c.u[0] = *(const uint4*)(p + kb);
    c.u[1] = *(const uint4*)(p + 16 + kb);
    return c.v;
}

// ---------------------------------------------------------------------------
// 1) router scores: one wave per row
// ---------------------------------------------------------------------------
__global__ __launch_bounds__(256) void scores_kernel(
    const float* __restrict__ x, const float* __restrict__ wsamp,
    const float* __restrict__ bsamp, float* __restrict__ scores) {
    int row  = blockIdx.x * 8 + (threadIdx.x >> 5);
    int lane = threadIdx.x & 31;
    const float* xr = x + (size_t)row * Dc;
    float s = 0.0f;
    for (int d = lane; d < Dc; d += 32) s += xr[d] * wsamp[d];
#pragma unroll
    for (int o = 16; o > 0; o >>= 1) s += __shfl_down(s, o);
    if (lane == 0) scores[row] = s + bsamp[0];
}

__global__ void init_pos_kernel(int* __restrict__ pos, int n) {
    int i = blockIdx.x * 256 + threadIdx.x;
    if (i < n) pos[i] = -1;
}

// ---------------------------------------------------------------------------
// 2) per-(b,segment) top-256 of 2048 via bitonic sort, then ascending index
// ---------------------------------------------------------------------------
__global__ __launch_bounds__(1024) void topk_kernel(
    const float* __restrict__ scores, int* __restrict__ idx,
    int* __restrict__ pos) {
    __shared__ float sv[FSEG];
    __shared__ int   si[FSEG];
    int b   = blockIdx.x / NSF;
    int seg = blockIdx.x % NSF;
    int tid = threadIdx.x;
    const float* sc = scores + (size_t)b * Sc + (size_t)seg * FSEG;
    for (int i = tid; i < FSEG; i += 1024) { sv[i] = sc[i]; si[i] = i; }
    __syncthreads();
    for (int k = 2; k <= FSEG; k <<= 1) {              // descending by value
        for (int j = k >> 1; j > 0; j >>= 1) {
            for (int t = tid; t < FSEG; t += 1024) {
                int ixj = t ^ j;
                if (ixj > t) {
                    bool desc = ((t & k) == 0);
                    float a = sv[t], c = sv[ixj];
                    bool sw = desc ? (a < c) : (a > c);
                    if (sw) {
                        sv[t] = c; sv[ixj] = a;
                        int tmp = si[t]; si[t] = si[ixj]; si[ixj] = tmp;
                    }
                }
            }
            __syncthreads();
        }
    }
    for (int k = 2; k <= SEGc; k <<= 1) {              // ascending index sort
        for (int j = k >> 1; j > 0; j >>= 1) {
            if (tid < SEGc) {
                int ixj = tid ^ j;
                if (ixj > tid) {
                    bool asc = ((tid & k) == 0);
                    int a = si[tid], c = si[ixj];
                    bool sw = asc ? (a > c) : (a < c);
                    if (sw) { si[tid] = c; si[ixj] = a; }
                }
            }
            __syncthreads();
        }
    }
    if (tid < SEGc) {
        int tok = seg * FSEG + si[tid];
        int jg  = seg * SEGc + tid;
        idx[b * RLc + jg] = tok;
        pos[b * Sc + tok] = jg;
    }
}

// ---------------------------------------------------------------------------
// 3) gather routed tokens, fp32 -> bf16
// ---------------------------------------------------------------------------
__global__ __launch_bounds__(256) void gather_kernel(
    const float* __restrict__ x, const int* __restrict__ idx,
    __bf16* __restrict__ xsel) {
    int r = blockIdx.x;
    int b = r / RLc;
    int j = r % RLc;
    int tok = idx[b * RLc + j];
    const float* src = x + ((size_t)b * Sc + tok) * Dc;
    __bf16* dst = xsel + (size_t)r * Dc;
    for (int d = threadIdx.x; d < Dc; d += 256) dst[d] = f2bf(src[d]);
}

// ---------------------------------------------------------------------------
// 4) WMMA GEMM: C[M,N] = act(A[M,K] x Bw[K,N] + bias)
//    A bf16 row-major; Bw fp32 row-major (L2-resident) converted to bf16 into
//    a TRANSPOSED LDS tile so A- and B-fragments use the same contiguous
//    2x ds_load_b128 loader.  Block tile 128x128, BK=32, double-buffered LDS,
//    8 waves x (2x4 tiles) = 8 WMMA per wave per K-step, 1 barrier per step.
// ---------------------------------------------------------------------------
template <int EPI, bool OUTF32>
__global__ __launch_bounds__(256) void gemm_kernel(
    const __bf16* __restrict__ A, const float* __restrict__ Bw,
    const float* __restrict__ bias, void* __restrict__ Cout,
    int M, int N, int K) {
    __shared__ __bf16 As[2][128 * 32];
    __shared__ __bf16 Bt[2][128 * 32];   // Bt[n][k] = B[k][n]
    int tid  = threadIdx.x;
    int bm   = blockIdx.y * 128;
    int bn   = blockIdx.x * 128;
    int wave = tid >> 5, lane = tid & 31;
    int wm = (wave >> 1) * 32;           // 0,32,64,96
    int wn = (wave & 1) * 64;            // 0,64
    v8f acc[2][4];
#pragma unroll
    for (int i = 0; i < 2; ++i)
#pragma unroll
        for (int j = 0; j < 4; ++j) acc[i][j] = vzero();

    auto stage = [&](int buf, int k0) {
        {   // A: 128x32 bf16, 2x uint4 per thread
            int row = tid >> 1, hf = (tid & 1) * 16;
            const uint4* src = (const uint4*)(A + (size_t)(bm + row) * K + k0 + hf);
            uint4* dst = (uint4*)(&As[buf][row * 32 + hf]);
            dst[0] = src[0];
            dst[1] = src[1];
        }
        {   // B: 32x128 fp32 -> bf16, transposed into Bt
            int r = tid >> 3, c0 = (tid & 7) * 16;
            const float* src = Bw + (size_t)(k0 + r) * N + bn + c0;
#pragma unroll
            for (int i = 0; i < 16; ++i) Bt[buf][(c0 + i) * 32 + r] = f2bf(src[i]);
        }
    };

    int nsteps = K >> 5;
    stage(0, 0);
    __syncthreads();
    for (int it = 0; it < nsteps; ++it) {
        int cur = it & 1;
        if (it + 1 < nsteps) {
            __builtin_prefetch(A + (size_t)(bm + (tid >> 1)) * K + (it + 1) * 32, 0, 0);
            stage(cur ^ 1, (it + 1) * 32);
        }
        v16bf af0 = frag32(&As[cur][0], wm);
        v16bf af1 = frag32(&As[cur][0], wm + 16);
#pragma unroll
        for (int j = 0; j < 4; ++j) {
            v16bf bfj = frag32(&Bt[cur][0], wn + j * 16);
            acc[0][j] = wmma_bf16(af0, bfj, acc[0][j]);
            acc[1][j] = wmma_bf16(af1, bfj, acc[1][j]);
        }
        __syncthreads();
    }
    int n_in = lane & 15, half = lane >> 4;
#pragma unroll
    for (int i = 0; i < 2; ++i)
#pragma unroll
        for (int j = 0; j < 4; ++j)
#pragma unroll
            for (int r = 0; r < 8; ++r) {
                int gm = bm + wm + i * 16 + r + 8 * half;
                int gn = bn + wn + j * 16 + n_in;
                float v = acc[i][j][r];
                if (EPI >= 1) v += bias[gn];
                if (EPI == 1) v = 0.5f * v * (1.0f + erff(v * 0.70710678118f));
                if (OUTF32)
                    ((float*)Cout)[(size_t)gm * N + gn] = v;
                else
                    ((__bf16*)Cout)[(size_t)gm * N + gn] = f2bf(v);
            }
}

// ---------------------------------------------------------------------------
// 5) compressive-memory attention. One block per (b,h).
//    K staged row-major (q k^T fragments contiguous); V staged TRANSPOSED
//    (P.V and sig_k^T.V fragments contiguous). 128x128 f32 memory carried in
//    global ws (L2-resident, single-owner), z in LDS.
// ---------------------------------------------------------------------------
__global__ __launch_bounds__(256) void attn_kernel(
    const __bf16* __restrict__ qg, const __bf16* __restrict__ kg,
    const __bf16* __restrict__ vg, const float* __restrict__ betas,
    float* __restrict__ mem_all, __bf16* __restrict__ att) {
    __shared__ __bf16 Ks[SEGc * 128];       // 64 KB, Ks[j][dk]
    __shared__ __bf16 Vt[128 * SEGc];       // 64 KB, Vt[dv][j]
    __shared__ float  Pw[8][16 * 32];       // 16 KB per-wave S/P scratch
    __shared__ float  rmax[8][16], rsum[8][16], rcorr[8][16], denw[8][16];
    __shared__ float  zs[128];

    int tid  = threadIdx.x;
    int wave = tid >> 5, lane = tid & 31;
    int b = blockIdx.x / Hc;
    int h = blockIdx.x % Hc;
    float* memg = mem_all + (size_t)(b * Hc + h) * (DKc * DVc);

    for (int i = tid; i < DKc * DVc; i += 256) memg[i] = 0.0f;
    if (tid < 128) zs[tid] = 1.0f / (float)DKc;
    __threadfence();
    __syncthreads();

    const float scale = 0.08838834764831845f;  // 1/sqrt(128)

    for (int seg = 0; seg < NSF; ++seg) {
        int base = seg * SEGc;
        {   // stage K (row-major) and V (transposed)
            int j = tid;
            size_t gro = ((size_t)(b * RLc + base + j)) * Dc + h * 128;
            const uint4* ksrc = (const uint4*)(kg + gro);
            const uint4* vsrc = (const uint4*)(vg + gro);
            uint4* kdst = (uint4*)(Ks + j * 128);
            uint4 vreg[16];
#pragma unroll
            for (int t = 0; t < 16; ++t) { kdst[t] = ksrc[t]; vreg[t] = vsrc[t]; }
            const __bf16* vr = (const __bf16*)vreg;
            for (int d = 0; d < 128; ++d) Vt[d * 256 + j] = vr[d];
        }
        __syncthreads();

        // ---- retrieval + causal in-segment attention, per 16-row block
        for (int rb = wave; rb < 16; rb += 8) {
            int i0 = rb * 16;
            size_t qrow = (size_t)(b * RLc + base + i0);
            int m = lane & 15, kbk = (lane >> 4) << 3, hf = lane >> 4;
            // q fragments: raw bf16 bit-copy; sig_q: elementwise elu(q)+1
            v16bf qa[4], sqa[4];
#pragma unroll
            for (int t = 0; t < 4; ++t) {
                const __bf16* p = qg + (qrow + m) * Dc + h * 128 + t * 32;
                frag_u cq;
                cq.u[0] = *(const uint4*)(p + kbk);
                cq.u[1] = *(const uint4*)(p + 16 + kbk);
                qa[t] = cq.v;
                v16bf as;
#pragma unroll
                for (int e = 0; e < 16; ++e) as[e] = f2bf(elu1(bf2f(cq.v[e])));
                sqa[t] = as;
            }
            // denominator: sig_q . z
            {
                int r2 = lane & 15;
                const __bf16* qp = qg + (qrow + r2) * Dc + h * 128;
                float ds = 0.0f;
                for (int d = hf * 64; d < hf * 64 + 64; ++d)
                    ds += elu1(bf2f(qp[d])) * zs[d];
                ds += __shfl_xor(ds, 16);
                denw[wave][r2] = ds;
            }
            if (lane < 16) { rmax[wave][lane] = -3.0e38f; rsum[wave][lane] = 0.0f; }
            v8f o[8];
#pragma unroll
            for (int nt = 0; nt < 8; ++nt) o[nt] = vzero();

            int jc_max = (i0 + 15) / 32;
            for (int jc = 0; jc <= jc_max; ++jc) {
                // S tile (16x32) = q . k^T
#pragma unroll
                for (int s = 0; s < 2; ++s) {
                    v8f sc = vzero();
#pragma unroll
                    for (int t = 0; t < 4; ++t)
                        sc = wmma_bf16(qa[t], fragKT(Ks, jc * 32 + s * 16, t * 32), sc);
                    int n = lane & 15;
#pragma unroll
                    for (int r = 0; r < 8; ++r) {
                        int mrow = r + 8 * hf;
                        int ng = jc * 32 + s * 16 + n;
                        float v = sc[r] * scale;
                        if (ng > i0 + mrow) v = -3.0e38f;
                        Pw[wave][mrow * 32 + s * 16 + n] = v;
                    }
                }
                // online softmax over the 32-wide chunk
                {
                    int r2 = lane & 15;
                    float mx = -3.0e38f;
                    for (int c = 0; c < 16; ++c)
                        mx = fmaxf(mx, Pw[wave][r2 * 32 + hf * 16 + c]);
                    mx = fmaxf(mx, __shfl_xor(mx, 16));
                    float om = rmax[wave][r2];
                    float nm = fmaxf(om, mx);
                    float corr = __expf(om - nm);
                    float se = 0.0f;
                    for (int c = 0; c < 16; ++c) {
                        float* p = &Pw[wave][r2 * 32 + hf * 16 + c];
                        float v = *p;
                        float e = (v <= -1.0e37f) ? 0.0f : __expf(v - nm);
                        *p = e; se += e;
                    }
                    se += __shfl_xor(se, 16);
                    rmax[wave][r2]  = nm;
                    rsum[wave][r2]  = rsum[wave][r2] * corr + se;
                    rcorr[wave][r2] = corr;
                }
                // P fragment (fp32 scratch -> bf16), rescale O, accumulate P.V
                v16bf pa;
                {
                    const float* P = &Pw[wave][0];
#pragma unroll
                    for (int e = 0; e < 8; ++e) {
                        pa[e]     = f2bf(P[m * 32 + kbk + e]);
                        pa[e + 8] = f2bf(P[m * 32 + 16 + kbk + e]);
                    }
                }
                float cf[8];
#pragma unroll
                for (int r = 0; r < 8; ++r) cf[r] = rcorr[wave][r + 8 * hf];
#pragma unroll
                for (int nt = 0; nt < 8; ++nt) {
#pragma unroll
                    for (int r = 0; r < 8; ++r) o[nt][r] *= cf[r];
                    o[nt] = wmma_bf16(pa, frag256(Vt, nt * 16, jc * 32), o[nt]);
                }
            }
            // combine: gate * (sig_q.M / den) + (1-gate) * softmax(qk).V
            float invl[8], den[8];
#pragma unroll
            for (int r = 0; r < 8; ++r) {
                invl[r] = 1.0f / rsum[wave][r + 8 * hf];
                den[r]  = denw[wave][r + 8 * hf];
            }
#pragma unroll
            for (int nt = 0; nt < 8; ++nt) {
                v8f am = vzero();
#pragma unroll
                for (int t = 0; t < 4; ++t) {
                    int n = lane & 15;
                    v16bf bm;
#pragma unroll
                    for (int e = 0; e < 8; ++e) {
                        bm[e]     = f2bf(memg[(t * 32 + kbk + e) * 128 + nt * 16 + n]);
                        bm[e + 8] = f2bf(memg[(t * 32 + 16 + kbk + e) * 128 + nt * 16 + n]);
                    }
                    am = wmma_bf16(sqa[t], bm, am);
                }
                int dv = nt * 16 + (lane & 15);
                float g = 1.0f / (1.0f + __expf(-betas[h * DVc + dv]));
#pragma unroll
                for (int r = 0; r < 8; ++r) {
                    int mrow = r + 8 * hf;
                    float v = g * (am[r] / den[r]) + (1.0f - g) * (o[nt][r] * invl[r]);
                    att[(qrow + mrow) * Dc + h * 128 + dv] = f2bf(v);
                }
            }
        }
        __syncthreads();

        // ---- memory + z update (after retrieval; reference order)
        for (int d = tid; d < 128; d += 256) {
            float acc = 0.0f;
            for (int j = 0; j < SEGc; ++j) acc += elu1(bf2f(Ks[j * 128 + d]));
            zs[d] += acc;
        }
        {
            int dm = wave;                      // 8 waves x 8 tiles of 16x16
            int m = lane & 15, kbk = (lane >> 4) << 3, hf = lane >> 4;
            for (int dn = 0; dn < 8; ++dn) {
                v8f acc = vzero();
                for (int jc = 0; jc < 8; ++jc) {
                    v16bf ak;                   // A[d][j] = elu1(K[j][d])
#pragma unroll
                    for (int e = 0; e < 8; ++e) {
                        ak[e]     = f2bf(elu1(bf2f(Ks[(jc * 32 + kbk + e) * 128 + dm * 16 + m])));
                        ak[e + 8] = f2bf(elu1(bf2f(Ks[(jc * 32 + 16 + kbk + e) * 128 + dm * 16 + m])));
                    }
                    acc = wmma_bf16(ak, frag256(Vt, dn * 16, jc * 32), acc);
                }
                int n = lane & 15;
#pragma unroll
                for (int r = 0; r < 8; ++r) {
                    int d = dm * 16 + r + 8 * hf;
                    memg[d * 128 + dn * 16 + n] += acc[r];
                }
            }
        }
        __threadfence();
        __syncthreads();
    }
}

// ---------------------------------------------------------------------------
// 6) scatter-add routed MLP output + LayerNorm + mask
// ---------------------------------------------------------------------------
__global__ __launch_bounds__(256) void ln_kernel(
    const float* __restrict__ x, const float* __restrict__ y2,
    const int* __restrict__ pos, const float* __restrict__ lnw,
    const float* __restrict__ lnb, float* __restrict__ out,
    float* __restrict__ mask_out) {
    __shared__ float red[16];
    int row  = blockIdx.x;
    int b    = row / Sc;
    int tid  = threadIdx.x;
    int wave = tid >> 5, lane = tid & 31;
    int p    = pos[row];
    const float* xr = x + (size_t)row * Dc;
    const float* yr = (p >= 0) ? y2 + ((size_t)(b * RLc + p)) * Dc : nullptr;
    float vals[8];
    float sum = 0.0f, sq = 0.0f;
#pragma unroll
    for (int i = 0; i < 8; ++i) {
        int d = tid + i * 256;
        float v = xr[d] + ((p >= 0) ? yr[d] : 0.0f);
        vals[i] = v; sum += v; sq += v * v;
    }
#pragma unroll
    for (int o = 16; o > 0; o >>= 1) {
        sum += __shfl_down(sum, o);
        sq  += __shfl_down(sq, o);
    }
    if (lane == 0) { red[wave] = sum; red[8 + wave] = sq; }
    __syncthreads();
    float ts = 0.0f, tq = 0.0f;
#pragma unroll
    for (int w = 0; w < 8; ++w) { ts += red[w]; tq += red[8 + w]; }
    float mu   = ts * (1.0f / Dc);
    float var  = tq * (1.0f / Dc) - mu * mu;
    float rstd = rsqrtf(var + 1e-5f);
#pragma unroll
    for (int i = 0; i < 8; ++i) {
        int d = tid + i * 256;
        out[(size_t)row * Dc + d] = (vals[i] - mu) * rstd * lnw[d] + lnb[d];
    }
    if (tid == 0) mask_out[row] = (p >= 0) ? 1.0f : 0.0f;
}

// ---------------------------------------------------------------------------
// launch
// ---------------------------------------------------------------------------
extern "C" void kernel_launch(void* const* d_in, const int* in_sizes, int n_in,
                              void* d_out, int out_size, void* d_ws,
                              size_t ws_size, hipStream_t stream) {
    (void)in_sizes; (void)n_in; (void)out_size; (void)ws_size;
    const float* x     = (const float*)d_in[0];
    const float* wq    = (const float*)d_in[1];
    const float* wk    = (const float*)d_in[2];
    const float* wv    = (const float*)d_in[3];
    const float* betas = (const float*)d_in[4];
    const float* wo    = (const float*)d_in[5];
    const float* w1    = (const float*)d_in[6];
    const float* b1    = (const float*)d_in[7];
    const float* w2    = (const float*)d_in[8];
    const float* b2    = (const float*)d_in[9];
    const float* lnw   = (const float*)d_in[10];
    const float* lnb   = (const float*)d_in[11];
    const float* wsamp = (const float*)d_in[12];
    const float* bsamp = (const float*)d_in[13];

    float* out        = (float*)d_out;
    float* mask_out   = out + (size_t)Bc * Sc * Dc;
    float* scores_out = mask_out + (size_t)Bc * Sc;

    char* w = (char*)d_ws;
    size_t off = 0;
    auto alloc = [&](size_t bytes) {
        size_t cur = off;
        off += (bytes + 255) & ~(size_t)255;
        return cur;
    };
    int*    pos  = (int*)(w + alloc((size_t)Bc * Sc * 4));
    int*    idx  = (int*)(w + alloc((size_t)Bc * RLc * 4));
    __bf16* xsel = (__bf16*)(w + alloc((size_t)MSEL * Dc * 2));
    __bf16* qbuf = (__bf16*)(w + alloc((size_t)MSEL * Dc * 2));
    __bf16* kbuf = (__bf16*)(w + alloc((size_t)MSEL * Dc * 2));
    __bf16* vbuf = (__bf16*)(w + alloc((size_t)MSEL * Dc * 2));
    __bf16* attb = (__bf16*)(w + alloc((size_t)MSEL * Dc * 2));
    __bf16* ybuf = (__bf16*)(w + alloc((size_t)MSEL * Dc * 2));
    __bf16* hbuf = (__bf16*)(w + alloc((size_t)MSEL * DHc * 2));
    float*  y2   = (float*)(w + alloc((size_t)MSEL * Dc * 4));
    float*  memg = (float*)(w + alloc((size_t)Bc * Hc * DKc * DVc * 4));

    scores_kernel<<<Bc * Sc / 8, 256, 0, stream>>>(x, wsamp, bsamp, scores_out);
    init_pos_kernel<<<(Bc * Sc + 255) / 256, 256, 0, stream>>>(pos, Bc * Sc);
    topk_kernel<<<Bc * NSF, 1024, 0, stream>>>(scores_out, idx, pos);
    gather_kernel<<<MSEL, 256, 0, stream>>>(x, idx, xsel);

    dim3 g1(Dc / 128, MSEL / 128);
    gemm_kernel<0, false><<<g1, 256, 0, stream>>>(xsel, wq, nullptr, qbuf, MSEL, Dc, Dc);
    gemm_kernel<0, false><<<g1, 256, 0, stream>>>(xsel, wk, nullptr, kbuf, MSEL, Dc, Dc);
    gemm_kernel<0, false><<<g1, 256, 0, stream>>>(xsel, wv, nullptr, vbuf, MSEL, Dc, Dc);

    attn_kernel<<<Bc * Hc, 256, 0, stream>>>(qbuf, kbuf, vbuf, betas, memg, attb);

    gemm_kernel<0, false><<<g1, 256, 0, stream>>>(attb, wo, nullptr, ybuf, MSEL, Dc, Dc);
    dim3 g2(DHc / 128, MSEL / 128);
    gemm_kernel<1, false><<<g2, 256, 0, stream>>>(ybuf, w1, b1, hbuf, MSEL, DHc, Dc);
    gemm_kernel<2, true><<<g1, 256, 0, stream>>>(hbuf, w2, b2, y2, MSEL, Dc, DHc);

    ln_kernel<<<Bc * Sc, 256, 0, stream>>>(x, y2, pos, lnw, lnb, out, mask_out);
}
